// CrossEntropyPAUCLoss_42305427866232
// MI455X (gfx1250) — compile-verified, matrix operand order checked by reading, not run on hardware
//
#include <hip/hip_runtime.h>
#include <hip/hip_bf16.h>
#include <math.h>

typedef __attribute__((ext_vector_type(16))) _Float16 v16h;
typedef __attribute__((ext_vector_type(8)))  float    v8f;

#define NROWS 262144
#define NCLS  100
#define NBINS 4096
#define R0f   0.95f
#define R1f   1.00f
#define LAMf  0.5f
#define LSf   0.1f
#define MAXPAUCf 0.05f
#define LOGP_RANGE 32.0f

#define HIST_WORDS (NCLS*NBINS*2)
#define CE_OFF   HIST_WORDS
#define PAUC_OFF (HIST_WORDS+1)
#define WS_WORDS (HIST_WORDS+2)

// ---------------------------------------------------------------- zero ws
__global__ void zero_ws_kernel(unsigned int* __restrict__ ws) {
    int i = blockIdx.x * blockDim.x + threadIdx.x;
    if (i < WS_WORDS) ws[i] = 0u;
}

// ------------------------------------------- softmax + CE + ROC histogram
// One wave32 handles 16 rows. Lane l covers row (l&15); the two half-lanes
// (l<16, l>=16) split the 128-padded class axis. Row sums of exp() are done
// with v_wmma_f32_16x16x32_f16 against an all-ones B (f32 accumulate).
__global__ __launch_bounds__(256, 2)
void softmax_ce_hist_kernel(const float* __restrict__ pred,
                            const int*   __restrict__ targ,
                            const float* __restrict__ weight,
                            unsigned int* __restrict__ hist,
                            float* __restrict__ ce_accum)
{
    const int lane = threadIdx.x & 31;
    const int wv   = threadIdx.x >> 5;
    const int r    = lane & 15;
    const int half = lane >> 4;
    const int rowBase = blockIdx.x * 128 + wv * 16;
    const int row  = rowBase + r;

    // prefetch the next block's tile of predictions (global_prefetch_b8)
    if (blockIdx.x + 1 < gridDim.x)
        __builtin_prefetch(pred + (size_t)(rowBase + 128) * NCLS, 0, 1);

    // this lane's 64 candidate class slots: c = chunk*32 + half*16 + i
    const float* rp = pred + (size_t)row * NCLS;
    float x[64];
#pragma unroll
    for (int k = 0; k < 64; k++) {
        int c = (k >> 4) * 32 + half * 16 + (k & 15);
        x[k] = (c < NCLS) ? rp[c] : -1e30f;
    }

    // row max (combine the two half-lanes of this row)
    float m = x[0];
#pragma unroll
    for (int k = 1; k < 64; k++) m = fmaxf(m, x[k]);
    m = fmaxf(m, __shfl_xor(m, 16, 32));

    // sum(exp) via WMMA: D = exp(A - m) x ones ; every column of D = row sum
    v16h bones;
#pragma unroll
    for (int i = 0; i < 16; i++) bones[i] = (_Float16)1.0f;
    v8f acc = {};
#pragma unroll
    for (int ch = 0; ch < 4; ch++) {
        v16h a;
#pragma unroll
        for (int i = 0; i < 16; i++) {
            int k = ch * 16 + i;
            int c = ch * 32 + half * 16 + i;
            a[i] = (c < NCLS) ? (_Float16)__expf(x[k] - m) : (_Float16)0.0f;
        }
        acc = __builtin_amdgcn_wmma_f32_16x16x32_f16(
                  false, a, false, bones, (short)0, acc, false, false);
    }

    // C/D layout: VGPR j holds M=j (lanes 0-15) and M=j+8 (lanes 16-31)
    float se[16];
#pragma unroll
    for (int j = 0; j < 8; j++) {
        se[j]     = __shfl(acc[j], 0,  32);
        se[j + 8] = __shfl(acc[j], 16, 32);
    }
    float sumexp = se[0];
#pragma unroll
    for (int j = 1; j < 16; j++) sumexp = (r == j) ? se[j] : sumexp;
    float lse = __logf(sumexp);

    int t = targ[row];

    // binned ROC histogram on log-prob (monotonic transform of score)
#pragma unroll
    for (int k = 0; k < 64; k++) {
        int c = (k >> 4) * 32 + half * 16 + (k & 15);
        if (c < NCLS) {
            float logp = x[k] - m - lse;
            float key  = (logp + LOGP_RANGE) * (1.0f / LOGP_RANGE);
            key = fminf(fmaxf(key, 0.0f), 1.0f);
            int bin = (int)(key * (float)NBINS);
            bin = (bin < NBINS - 1) ? bin : (NBINS - 1);
            bin = (bin > 0) ? bin : 0;
            int isPos = (c == t) ? 1 : 0;
            atomicAdd(&hist[(c * NBINS + bin) * 2 + isPos], 1u);
        }
    }

    // weighted sums for label-smoothed CE (exact f32 path)
    float swx = 0.0f, sw = 0.0f;
#pragma unroll
    for (int k = 0; k < 64; k++) {
        int c = (k >> 4) * 32 + half * 16 + (k & 15);
        if (c < NCLS) {
            float w = weight[c];
            swx += x[k] * w;
            sw  += w;
        }
    }
    swx += __shfl_xor(swx, 16, 32);
    sw  += __shfl_xor(sw,  16, 32);

    float cev = 0.0f;
    if (half == 0) {
        float wt   = weight[t];
        float lp_t = rp[t] - m - lse;
        cev = -((1.0f - LSf) * wt * lp_t
                + (LSf / (float)NCLS) * (swx - (m + lse) * sw));
    }
#pragma unroll
    for (int off = 16; off >= 1; off >>= 1) cev += __shfl_xor(cev, off, 32);
    if (lane == 0) atomicAdd(ce_accum, cev);
}

// ------------------------------------------------ per-class pAUC integral
__global__ __launch_bounds__(256)
void pauc_kernel(const unsigned int* __restrict__ hist,
                 const float* __restrict__ weight,
                 float* __restrict__ pauc_accum)
{
    const int c   = blockIdx.x;
    const int tid = threadIdx.x;
    const int ITEMS = NBINS / 256; // 16
    __shared__ float s_tp[256], s_fp[256];

    float pos[ITEMS], neg[ITEMS];
    float tpsum = 0.0f, fpsum = 0.0f;
#pragma unroll
    for (int i = 0; i < ITEMS; i++) {
        int d   = tid * ITEMS + i;       // descending-score order
        int bin = NBINS - 1 - d;
        pos[i] = (float)hist[(c * NBINS + bin) * 2 + 1];
        neg[i] = (float)hist[(c * NBINS + bin) * 2 + 0];
        tpsum += pos[i]; fpsum += neg[i];
    }
    s_tp[tid] = tpsum; s_fp[tid] = fpsum; __syncthreads();
    for (int off = 1; off < 256; off <<= 1) {
        float a = 0.0f, b = 0.0f;
        if (tid >= off) { a = s_tp[tid - off]; b = s_fp[tid - off]; }
        __syncthreads();
        s_tp[tid] += a; s_fp[tid] += b;
        __syncthreads();
    }
    float P  = s_tp[255];
    float Nn = s_fp[255];
    float tp = s_tp[tid] - tpsum;    // exclusive prefix
    float fp = s_fp[tid] - fpsum;
    float invP  = (P  > 0.0f) ? 1.0f / P  : 0.0f;
    float invNn = (Nn > 0.0f) ? 1.0f / Nn : 0.0f;

    float contrib = 0.0f;
#pragma unroll
    for (int i = 0; i < ITEMS; i++) {
        float tp1 = tp + pos[i];
        float fp1 = fp + neg[i];
        float tpr0 = tp  * invP;
        float tpr1 = tp1 * invP;
        if (tpr0 >= R0f && tpr0 <= R1f && tpr1 >= R0f && tpr1 <= R1f)
            contrib += 0.5f * ((tpr0 - R0f) + (tpr1 - R0f)) * (fp1 - fp) * invNn;
        tp = tp1; fp = fp1;
    }
    __syncthreads();
    s_tp[tid] = contrib; __syncthreads();
    for (int off = 128; off >= 1; off >>= 1) {
        if (tid < off) s_tp[tid] += s_tp[tid + off];
        __syncthreads();
    }
    if (tid == 0) atomicAdd(pauc_accum, s_tp[0] * weight[c]);
}

// ----------------------------------------------------------------- final
__global__ void finalize_kernel(const float* __restrict__ weight,
                                const float* __restrict__ ce_accum,
                                const float* __restrict__ pauc_accum,
                                float* __restrict__ out)
{
    __shared__ float s[128];
    int tid = threadIdx.x;
    float w = 0.0f;
    for (int c = tid; c < NCLS; c += 128) w += weight[c];
    s[tid] = w; __syncthreads();
    for (int off = 64; off >= 1; off >>= 1) {
        if (tid < off) s[tid] += s[tid + off];
        __syncthreads();
    }
    if (tid == 0) {
        float sumw = s[0];
        float ce   = ce_accum[0] / (float)NROWS;
        float avg  = pauc_accum[0] / (sumw * MAXPAUCf);
        avg = fminf(fmaxf(avg, 0.0f), 1.0f);
        out[0] = (1.0f - LAMf) * ce + LAMf * (1.0f - avg * avg);
    }
}

extern "C" void kernel_launch(void* const* d_in, const int* in_sizes, int n_in,
                              void* d_out, int out_size, void* d_ws, size_t ws_size,
                              hipStream_t stream)
{
    (void)in_sizes; (void)n_in; (void)out_size; (void)ws_size;
    const float* pred   = (const float*)d_in[0];
    const int*   targ   = (const int*)  d_in[1];
    const float* weight = (const float*)d_in[2];

    unsigned int* ws   = (unsigned int*)d_ws;
    unsigned int* hist = ws;
    float* ce_acc   = (float*)(ws + CE_OFF);
    float* pauc_acc = (float*)(ws + PAUC_OFF);
    float* out      = (float*)d_out;

    zero_ws_kernel<<<(WS_WORDS + 255) / 256, 256, 0, stream>>>(ws);
    softmax_ce_hist_kernel<<<NROWS / 128, 256, 0, stream>>>(pred, targ, weight,
                                                            hist, ce_acc);
    pauc_kernel<<<NCLS, 256, 0, stream>>>(hist, weight, pauc_acc);
    finalize_kernel<<<1, 128, 0, stream>>>(weight, ce_acc, pauc_acc, out);
}
